// Mamba_20392504721968
// MI455X (gfx1250) — matrix operand order repaired and checked by reading, not compiled, not run on hardware
//
#include <hip/hip_runtime.h>
#include <cstdint>

// ---------------- problem constants ----------------
#define BB   32
#define TT   20
#define NVV  30
#define CC   64
#define DMAMB  1280
#define DTRANS 1920
#define DINNER 2560
#define DSTATE 16
#define DTRANK 80
#define TEMPD  512
#define FFND   2560
#define NHEADS 8
#define DHEAD  240
#define NTOK3  (BB*TT*NVV*CC)   // 1,228,800

typedef __bf16 bf16_t;
typedef __attribute__((ext_vector_type(16))) __bf16 v16bf;
typedef __attribute__((ext_vector_type(8)))  __bf16 v8bf;
typedef __attribute__((ext_vector_type(4)))  __bf16 v4bf;
typedef __attribute__((ext_vector_type(8)))  float  v8f;
typedef __attribute__((ext_vector_type(4)))  float  f4v;

__constant__ int c_VORDER[NVV] = {9,8,7,10,11,12,11,10,7,13,14,15,14,13,7,6,
                                  0,1,2,1,0,6,3,4,5,4,3,6,7,8};
__constant__ int c_REV[16]     = {20,19,18,26,25,24,27,28,29,0,7,6,5,13,12,11};

__device__ __forceinline__ float siluf(float x){ return x / (1.f + __expf(-x)); }
__device__ __forceinline__ float geluf(float x){
  float x3 = x*x*x;
  return 0.5f*x*(1.f + tanhf(0.7978845608028654f*(x + 0.044715f*x3)));
}
__device__ __forceinline__ float softplusf(float x){
  return (x > 20.f) ? x : log1pf(__expf(x));
}

// ---------------- WMMA GEMM:  C = act(A[MxK] * W[KxN] + bias) (+ res) --------
// Double-buffered LDS staging; fragment loads issued before staging stores so
// the WMMA stream's dscnt waits exclude them; staging itself is two-phase
// (batch all global_load_b128 into distinct regs, then cvt+ds_store) so the
// 8 loads pipeline instead of serializing on s_wait_loadcnt 0.
enum { EPI_NONE = 0, EPI_SILU = 1, EPI_GELU = 2, EPI_SOFTPLUS = 3 };

#define LDSP 40   // padded LDS row stride (bf16 elems): 20 dwords, conflict-free

template<int EPI, bool HAS_RES, bool KGUARD, bool NFULL>
__global__ __launch_bounds__(128)
void gemm_bf16_wmma(const float* __restrict__ A, int lda,
                    const float* __restrict__ W,
                    const float* __restrict__ bias,
                    const float* __restrict__ res, int ldr,
                    float* __restrict__ C, int ldc,
                    int M, int N, int K)
{
  __shared__ __bf16 lA[2][64*LDSP];   // [m][k] 64x32 (+pad)
  __shared__ __bf16 lB[2][64*LDSP];   // [n][k] 64x32 (+pad), transposed tile
  const int tid  = threadIdx.x;
  const int lane = tid & 31;
  const int wave = tid >> 5;                 // 4 waves / block
  const int rowblk = blockIdx.y * 64;
  const int row0 = rowblk + wave * 16;
  const int colb = blockIdx.x * 64;
  const int hl = lane >> 4;                  // half-wave select
  const int ln = lane & 15;

  auto stage = [&](__bf16* dA, __bf16* dB, int kb) {
    f4v va[4], vb[4];
    // phase 1: batch ALL global loads into distinct registers (pipelined)
#pragma unroll
    for (int i = 0; i < 4; ++i) {
      int f  = tid + i*128;                  // 0..511
      int r  = f >> 3;                       // tile row 0..63
      int kq = (f & 7) << 2;                 // k offset 0..28
      int grow = rowblk + r;
      if (grow >= M) grow = M - 1;           // clamp: rows >= M never stored
      int kk = kb + kq;
      if (KGUARD && kk > K - 4) kk = K - 4;  // clamp: B zeroes the K-tail
      va[i] = *(const f4v*)(A + (size_t)grow*lda + kk);
    }
#pragma unroll
    for (int i = 0; i < 4; ++i) {
      int f  = tid + i*128;                  // 0..511
      int kk = f >> 4;                       // 0..31
      int nq = (f & 15) << 2;                // 0..60
      int gk = kb + kk;
      int gn = colb + nq;
      bool kok = KGUARD ? (gk < K) : true;
      bool nok = NFULL ? true : (gn < N);
      if (kok && nok) vb[i] = *(const f4v*)(W + (size_t)gk*N + gn);
      else            vb[i] = (f4v){0.f,0.f,0.f,0.f};
    }
    // phase 2: convert + LDS stores (waits retire loads incrementally)
#pragma unroll
    for (int i = 0; i < 4; ++i) {
      int f  = tid + i*128;
      int r  = f >> 3;
      int kq = (f & 7) << 2;
      *(v4bf*)&dA[r*LDSP + kq] =
          (v4bf){(bf16_t)va[i][0], (bf16_t)va[i][1], (bf16_t)va[i][2], (bf16_t)va[i][3]};
    }
#pragma unroll
    for (int i = 0; i < 4; ++i) {
      int f  = tid + i*128;
      int kk = f >> 4;
      int nq = (f & 15) << 2;
      dB[(nq+0)*LDSP + kk] = (bf16_t)vb[i][0];
      dB[(nq+1)*LDSP + kk] = (bf16_t)vb[i][1];
      dB[(nq+2)*LDSP + kk] = (bf16_t)vb[i][2];
      dB[(nq+3)*LDSP + kk] = (bf16_t)vb[i][3];
    }
  };

  v8f acc[4] = {};

  stage(lA[0], lB[0], 0);
  __syncthreads();

  int buf = 0;
  for (int k0 = 0; k0 < K; k0 += 32) {
    int nxt = buf ^ 1;

    // 1) fragment loads from current buffer FIRST (so later waits on them
    //    do not also drain the staging stores issued below)
    const __bf16* ap = &lA[buf][(wave*16 + ln)*LDSP + 8*hl];
    v8bf a0 = *(const v8bf*)(ap);
    v8bf a1 = *(const v8bf*)(ap + 16);
    v8bf b0[4], b1[4];
#pragma unroll
    for (int nb = 0; nb < 4; ++nb) {
      if (!NFULL && (colb + nb*16 >= N)) break;
      const __bf16* bp = &lB[buf][(nb*16 + ln)*LDSP + 8*hl];
      b0[nb] = *(const v8bf*)(bp);
      b1[nb] = *(const v8bf*)(bp + 16);
    }

    // 2) stage next tile (global->LDS) while fragment loads are in flight
    if (k0 + 32 < K) {
      stage(lA[nxt], lB[nxt], k0 + 32);
      if (k0 + 64 < K) {                     // prefetch hint two tiles ahead
        int r  = tid >> 3;
        int kq = (tid & 7) << 2;
        int grow = rowblk + r;
        if (grow >= M) grow = M - 1;
        __builtin_prefetch(A + (size_t)grow*lda + k0 + 64 + kq, 0, 0);
      }
    }

    // 3) WMMA stream: A 16x32 bf16 lane layout per ISA 7.12.2
    v16bf af = __builtin_shufflevector(a0, a1, 0,1,2,3,4,5,6,7,8,9,10,11,12,13,14,15);
#pragma unroll
    for (int nb = 0; nb < 4; ++nb) {
      if (!NFULL && (colb + nb*16 >= N)) break;
      v16bf bfrag = __builtin_shufflevector(b0[nb], b1[nb],
                                            0,1,2,3,4,5,6,7,8,9,10,11,12,13,14,15);
      acc[nb] = __builtin_amdgcn_wmma_f32_16x16x32_bf16(
          false, af, false, bfrag, (short)0, acc[nb], false, false);
    }
    __syncthreads();
    buf = nxt;
  }

  // C/D layout: VGPR r -> row r (lanes 0-15) or 8+r (lanes 16-31); col = lane&15.
#pragma unroll
  for (int nb = 0; nb < 4; ++nb) {
    int col0 = colb + nb*16;
    if (!NFULL && col0 >= N) break;
    int col = col0 + ln;
    float bv = bias ? bias[col] : 0.f;
#pragma unroll
    for (int r = 0; r < 8; ++r) {
      int row = row0 + hl*8 + r;
      if (row < M) {
        float v = acc[nb][r] + bv;
        if (EPI == EPI_SILU)          v = siluf(v);
        else if (EPI == EPI_GELU)     v = geluf(v);
        else if (EPI == EPI_SOFTPLUS) v = softplusf(v);
        if (HAS_RES) v += res[(size_t)row*ldr + col];
        C[(size_t)row*ldc + col] = v;
      }
    }
  }
}

template<int EPI, bool HAS_RES, bool KGUARD, bool NFULL>
static void launch_gemm(const float* A, int lda, const float* W, const float* bias,
                        const float* res, int ldr, float* C, int ldc,
                        int M, int N, int K, hipStream_t stream)
{
  dim3 grid((N + 63) / 64, (M + 63) / 64);
  gemm_bf16_wmma<EPI, HAS_RES, KGUARD, NFULL><<<grid, 128, 0, stream>>>(
      A, lda, W, bias, res, ldr, C, ldc, M, N, K);
}

// ---------------- small kernels ----------------
__global__ void k_timestep_emb(const int* __restrict__ ts, float* __restrict__ out)
{
  int b = blockIdx.x;
  int j = blockIdx.y * blockDim.x + threadIdx.x;
  if (j >= DMAMB) return;
  float t = (float)ts[b];
  int half = DMAMB / 2;
  int idx = (j < half) ? j : (j - half);
  float freq = __expf(-9.210340371976184f * (float)idx / (float)half); // ln(1e4)
  float a = t * freq;
  out[b*DMAMB + j] = (j < half) ? __cosf(a) : __sinf(a);
}

__global__ void k_silu_ew(const float* __restrict__ x, float* __restrict__ y, int n)
{
  int i = blockIdx.x*blockDim.x + threadIdx.x;
  if (i < n) y[i] = siluf(x[i]);
}

__global__ void k_add_ew(float* __restrict__ a, const float* __restrict__ b, int n)
{
  int i = blockIdx.x*blockDim.x + threadIdx.x;
  if (i < n) a[i] += b[i];
}

// h[b,t,vo,c] = sum_r x[b,t,VORDER[vo]*3+r]*emb_W[r,c] + emb_b[c] + seq_emb[t,VORDER[vo],c]
__global__ void k_embed_h(const float* __restrict__ x, const float* __restrict__ eW,
                          const float* __restrict__ eb, const float* __restrict__ se,
                          float* __restrict__ h)
{
  int idx = blockIdx.x*blockDim.x + threadIdx.x;
  if (idx >= NTOK3) return;
  int c  = idx & 63;
  int vo = (idx >> 6) % NVV;
  int t  = ((idx >> 6) / NVV) % TT;
  int b  = idx / (CC*NVV*TT);
  int v  = c_VORDER[vo];
  const float* xp = x + ((size_t)(b*TT + t)*48 + v*3);
  float acc = eb[c] + se[((size_t)t*16 + v)*CC + c];
  acc += xp[0]*eW[c] + xp[1]*eW[64 + c] + xp[2]*eW[128 + c];
  h[idx] = acc;
}

// hm[b,v,t*64+c] = h[b,t,v,c]
__global__ void k_t2hm(const float* __restrict__ h, float* __restrict__ hm)
{
  int idx = blockIdx.x*blockDim.x + threadIdx.x;
  if (idx >= NTOK3) return;
  int c = idx & 63;
  int t = (idx >> 6) % TT;
  int v = ((idx >> 6) / TT) % NVV;
  int b = idx / (CC*TT*NVV);
  hm[idx] = h[(((size_t)(b*TT + t)*NVV + v) << 6) + c];
}

// h[b,t,v,c] = hm[b,v,t*64+c]
__global__ void k_hm2h(const float* __restrict__ hm, float* __restrict__ h)
{
  int idx = blockIdx.x*blockDim.x + threadIdx.x;
  if (idx >= NTOK3) return;
  int c = idx & 63;
  int v = (idx >> 6) % NVV;
  int t = ((idx >> 6) / NVV) % TT;
  int b = idx / (CC*NVV*TT);
  h[idx] = hm[(((size_t)(b*NVV + v)*TT + t) << 6) + c];
}

__global__ __launch_bounds__(256)
void k_rmsnorm_add(const float* __restrict__ x, const float* __restrict__ w,
                   const float* __restrict__ add, float* __restrict__ out,
                   int width, int rows_per_b)
{
  int row = blockIdx.x, tid = threadIdx.x;
  const float* xr = x + (size_t)row*width;
  __shared__ float red[256];
  float s = 0.f;
  for (int j = tid; j < width; j += 256) { float v = xr[j]; s += v*v; }
  red[tid] = s; __syncthreads();
  for (int k = 128; k > 0; k >>= 1) { if (tid < k) red[tid] += red[tid+k]; __syncthreads(); }
  float rs = rsqrtf(red[0] / (float)width + 1e-5f);
  int b = row / rows_per_b;
  for (int j = tid; j < width; j += 256) {
    float v = xr[j] * rs * w[j];
    if (add) v += add[(size_t)b*width + j];
    out[(size_t)row*width + j] = v;
  }
}

__global__ __launch_bounds__(256)
void k_layernorm(const float* __restrict__ x, const float* __restrict__ w,
                 const float* __restrict__ bvec, float* __restrict__ out, int width)
{
  int row = blockIdx.x, tid = threadIdx.x;
  const float* xr = x + (size_t)row*width;
  __shared__ float r1[256], r2[256];
  float s1 = 0.f, s2 = 0.f;
  for (int j = tid; j < width; j += 256) { float v = xr[j]; s1 += v; s2 += v*v; }
  r1[tid] = s1; r2[tid] = s2; __syncthreads();
  for (int k = 128; k > 0; k >>= 1) {
    if (tid < k) { r1[tid] += r1[tid+k]; r2[tid] += r2[tid+k]; } __syncthreads();
  }
  float m  = r1[0] / (float)width;
  float var = r2[0] / (float)width - m*m;
  float rs = rsqrtf(var + 1e-5f);
  for (int j = tid; j < width; j += 256)
    out[(size_t)row*width + j] = (xr[j] - m) * rs * w[j] + bvec[j];
}

// out = silu( LN(x)*(1+scale[b]) + shift[b] ), scale|shift packed in ss[b, 2*width]
__global__ __launch_bounds__(256)
void k_ln_style(const float* __restrict__ x, const float* __restrict__ lw,
                const float* __restrict__ lb, const float* __restrict__ ss,
                float* __restrict__ out, int width)
{
  int row = blockIdx.x, tid = threadIdx.x;
  int b = row / TT;
  const float* xr = x + (size_t)row*width;
  __shared__ float r1[256], r2[256];
  float s1 = 0.f, s2 = 0.f;
  for (int j = tid; j < width; j += 256) { float v = xr[j]; s1 += v; s2 += v*v; }
  r1[tid] = s1; r2[tid] = s2; __syncthreads();
  for (int k = 128; k > 0; k >>= 1) {
    if (tid < k) { r1[tid] += r1[tid+k]; r2[tid] += r2[tid+k]; } __syncthreads();
  }
  float m  = r1[0] / (float)width;
  float var = r2[0] / (float)width - m*m;
  float rs = rsqrtf(var + 1e-5f);
  const float* sb = ss + (size_t)b * 2 * width;
  for (int j = tid; j < width; j += 256) {
    float ln = (xr[j] - m) * rs * lw[j] + lb[j];
    out[(size_t)row*width + j] = siluf(ln * (1.f + sb[j]) + sb[width + j]);
  }
}

// xc = silu(causal depthwise conv(xm)), xm = xz[:, :DINNER] with ld 2*DINNER
__global__ void k_conv_silu(const float* __restrict__ xz, const float* __restrict__ cw,
                            const float* __restrict__ cb, float* __restrict__ xc)
{
  int idx = blockIdx.x*blockDim.x + threadIdx.x;
  if (idx >= BB*NVV*DINNER) return;
  int d = idx % DINNER;
  int row = idx / DINNER;
  int b = row / NVV, l = row % NVV;
  float acc = cb[d];
#pragma unroll
  for (int k = 0; k < 4; ++k) {
    int ls = l + k - 3;
    if (ls >= 0) acc += xz[(size_t)(b*NVV + ls) * (2*DINNER) + d] * cw[d*4 + k];
  }
  xc[idx] = siluf(acc);
}

// selective scan over L=30; one thread per (b,d)
__global__ void k_ssm_scan(const float* __restrict__ delta, const float* __restrict__ dbc,
                           const float* __restrict__ xc, const float* __restrict__ xz,
                           const float* __restrict__ A_log, const float* __restrict__ Dp,
                           float* __restrict__ ymul)
{
  int idx = blockIdx.x*blockDim.x + threadIdx.x;
  if (idx >= BB*DINNER) return;
  int d = idx % DINNER, b = idx / DINNER;
  float Ad[DSTATE];
#pragma unroll
  for (int n = 0; n < DSTATE; ++n) Ad[n] = -__expf(A_log[(size_t)d*DSTATE + n]);
  float s[DSTATE];
#pragma unroll
  for (int n = 0; n < DSTATE; ++n) s[n] = 0.f;
  float Dpd = Dp[d];
  for (int l = 0; l < NVV; ++l) {
    size_t row = (size_t)b*NVV + l;
    float del = delta[row*DINNER + d];
    float xcv = xc[row*DINNER + d];
    const float* bc = dbc + row*112;
    float y = 0.f;
#pragma unroll
    for (int n = 0; n < DSTATE; ++n) {
      float dA = __expf(del * Ad[n]);
      s[n] = dA * s[n] + del * bc[DTRANK + n] * xcv;
      y += s[n] * bc[DTRANK + DSTATE + n];
    }
    y += xcv * Dpd;
    float r = xz[row*(2*DINNER) + DINNER + d];
    ymul[row*DINNER + d] = y * siluf(r);
  }
}

// attention: one block per (b, head); t = s = 20, dh = 240
__global__ __launch_bounds__(256)
void k_attention(const float* __restrict__ qg, const float* __restrict__ kg,
                 const float* __restrict__ vg, float* __restrict__ yg)
{
  int bh = blockIdx.x;
  int b = bh >> 3, hh = bh & 7;
  __shared__ float sq[TT*DHEAD], sk[TT*DHEAD], sv[TT*DHEAD], sc[TT*TT];
  for (int i = threadIdx.x; i < TT*DHEAD; i += 256) {
    int t = i / DHEAD, d = i % DHEAD;
    size_t g = (size_t)(b*TT + t)*DTRANS + hh*DHEAD + d;
    sq[i] = qg[g]; sk[i] = kg[g]; sv[i] = vg[g];
  }
  __syncthreads();
  for (int i = threadIdx.x; i < TT*TT; i += 256) {
    int t = i / TT, s = i % TT;
    float acc = 0.f;
    for (int d = 0; d < DHEAD; ++d) acc += sq[t*DHEAD + d]*sk[s*DHEAD + d];
    sc[i] = acc * 0.06454972243679028f;   // 1/sqrt(240)
  }
  __syncthreads();
  if (threadIdx.x < TT) {
    int t = threadIdx.x;
    float m = -1e30f;
    for (int s = 0; s < TT; ++s) m = fmaxf(m, sc[t*TT + s]);
    float sum = 0.f;
    for (int s = 0; s < TT; ++s) { float e = __expf(sc[t*TT + s] - m); sc[t*TT + s] = e; sum += e; }
    float inv = 1.f / sum;
    for (int s = 0; s < TT; ++s) sc[t*TT + s] *= inv;
  }
  __syncthreads();
  for (int i = threadIdx.x; i < TT*DHEAD; i += 256) {
    int t = i / DHEAD, d = i % DHEAD;
    float acc = 0.f;
    for (int s = 0; s < TT; ++s) acc += sc[t*TT + s]*sv[s*DHEAD + d];
    yg[(size_t)(b*TT + t)*DTRANS + hh*DHEAD + d] = acc;
  }
}

// final: out[b,t,v*3+r] = hf[b,t, REV[v]*64 : +64] . head_W[:,r] + head_b[r]
__global__ void k_head(const float* __restrict__ hf, const float* __restrict__ hW,
                       const float* __restrict__ hb, float* __restrict__ out)
{
  int idx = blockIdx.x*blockDim.x + threadIdx.x;
  if (idx >= BB*TT*16*3) return;
  int r  = idx % 3;
  int v  = (idx / 3) % 16;
  int bt = idx / 48;
  const float* hr = hf + (size_t)bt*DTRANS + c_REV[v]*CC;
  float acc = hb[r];
#pragma unroll
  for (int c = 0; c < CC; ++c) acc += hr[c]*hW[c*3 + r];
  out[idx] = acc;
}

// ---------------- orchestration ----------------
extern "C" void kernel_launch(void* const* d_in, const int* in_sizes, int n_in,
                              void* d_out, int out_size, void* d_ws, size_t ws_size,
                              hipStream_t stream)
{
  (void)in_sizes; (void)n_in; (void)out_size; (void)ws_size;
  const float* x        = (const float*)d_in[0];
  const int*   tsteps   = (const int*)  d_in[1];
  const float* seq_emb  = (const float*)d_in[2];
  const float* emb_W    = (const float*)d_in[3];
  const float* emb_b    = (const float*)d_in[4];
  const float* time_W1  = (const float*)d_in[5];
  const float* time_b1  = (const float*)d_in[6];
  const float* time_W2  = (const float*)d_in[7];
  const float* time_b2  = (const float*)d_in[8];
  const float* m_norm_w = (const float*)d_in[9];
  const float* m_emb_W  = (const float*)d_in[10];
  const float* m_emb_b  = (const float*)d_in[11];
  const float* m_in_W   = (const float*)d_in[12];
  const float* m_conv_w = (const float*)d_in[13];
  const float* m_conv_b = (const float*)d_in[14];
  const float* m_xproj  = (const float*)d_in[15];
  const float* m_dt_W   = (const float*)d_in[16];
  const float* m_dt_b   = (const float*)d_in[17];
  const float* m_A_log  = (const float*)d_in[18];
  const float* m_D      = (const float*)d_in[19];
  const float* m_out_W  = (const float*)d_in[20];
  const float* t_ln1_w  = (const float*)d_in[21];
  const float* t_ln1_b  = (const float*)d_in[22];
  const float* t_Wq     = (const float*)d_in[23];
  const float* t_bq     = (const float*)d_in[24];
  const float* t_Wk     = (const float*)d_in[25];
  const float* t_bk     = (const float*)d_in[26];
  const float* t_Wv     = (const float*)d_in[27];
  const float* t_bv     = (const float*)d_in[28];
  const float* t_s1_eW  = (const float*)d_in[29];
  const float* t_s1_eb  = (const float*)d_in[30];
  const float* t_s1_lw  = (const float*)d_in[31];
  const float* t_s1_lb  = (const float*)d_in[32];
  const float* t_s1_oW  = (const float*)d_in[33];
  const float* t_s1_ob  = (const float*)d_in[34];
  const float* t_ln2_w  = (const float*)d_in[35];
  const float* t_ln2_b  = (const float*)d_in[36];
  const float* t_fW1    = (const float*)d_in[37];
  const float* t_fb1    = (const float*)d_in[38];
  const float* t_fW2    = (const float*)d_in[39];
  const float* t_fb2    = (const float*)d_in[40];
  const float* t_s2_eW  = (const float*)d_in[41];
  const float* t_s2_eb  = (const float*)d_in[42];
  const float* t_s2_lw  = (const float*)d_in[43];
  const float* t_s2_lb  = (const float*)d_in[44];
  const float* t_s2_oW  = (const float*)d_in[45];
  const float* t_s2_ob  = (const float*)d_in[46];
  const float* normf_w  = (const float*)d_in[47];
  const float* head_W   = (const float*)d_in[48];
  const float* head_b   = (const float*)d_in[49];
  float* out = (float*)d_out;

  // ---- workspace carve (fp32 elements) ----
  float* ws = (float*)d_ws;
  size_t off = 0;
  auto take = [&](size_t n) { float* p = ws + off; off += n; return p; };
  float* temb0 = take((size_t)BB*DMAMB);
  float* e1    = take((size_t)BB*TEMPD);
  float* temb  = take((size_t)BB*TEMPD);
  float* stemb = take((size_t)BB*TEMPD);
  float* embm  = take((size_t)BB*DMAMB);
  float* hbuf  = take(NTOK3);
  float* pre0  = take(NTOK3);
  float* pre1  = take(NTOK3);
  float* hm    = take(NTOK3);
  float* hn    = take(NTOK3);
  float* xz    = take((size_t)BB*NVV*2*DINNER);
  float* xc    = take((size_t)BB*NVV*DINNER);
  float* dbc   = take((size_t)BB*NVV*112);
  float* delta = take((size_t)BB*NVV*DINNER);
  float* ymul  = take((size_t)BB*NVV*DINNER);
  float* xn    = take((size_t)BB*TT*DTRANS);
  float* qb    = take((size_t)BB*TT*DTRANS);
  float* kb    = take((size_t)BB*TT*DTRANS);
  float* vb    = take((size_t)BB*TT*DTRANS);
  float* yb    = take((size_t)BB*TT*DTRANS);
  float* ssb   = take((size_t)BB*2*DTRANS);
  float* f1    = take((size_t)BB*TT*FFND);
  float* sh2   = take((size_t)BB*TT*DTRANS);

  const int EB = 256;
  // 1) timestep embedding + time MLP
  k_timestep_emb<<<dim3(BB, (DMAMB+EB-1)/EB), EB, 0, stream>>>(tsteps, temb0);
  launch_gemm<EPI_SILU,false,false,true>(temb0, DMAMB, time_W1, time_b1, nullptr,0,
                                         e1, TEMPD, BB, TEMPD, DMAMB, stream);
  launch_gemm<EPI_NONE,false,false,true>(e1, TEMPD, time_W2, time_b2, nullptr,0,
                                         temb, TEMPD, BB, TEMPD, TEMPD, stream);
  k_silu_ew<<<(BB*TEMPD+EB-1)/EB, EB, 0, stream>>>(temb, stemb, BB*TEMPD);
  // 2) input embed + VORDER gather
  k_embed_h<<<(NTOK3+EB-1)/EB, EB, 0, stream>>>(x, emb_W, emb_b, seq_emb, hbuf);

  for (int i = 0; i < 4; ++i) {
    if (i == 0) hipMemcpyAsync(pre0, hbuf, (size_t)NTOK3*sizeof(float),
                               hipMemcpyDeviceToDevice, stream);
    if (i == 1) hipMemcpyAsync(pre1, hbuf, (size_t)NTOK3*sizeof(float),
                               hipMemcpyDeviceToDevice, stream);
    // ---- mamba branch (sequence dim = V=30, features = t*c = 1280) ----
    k_t2hm<<<(NTOK3+EB-1)/EB, EB, 0, stream>>>(hbuf, hm);
    launch_gemm<EPI_NONE,false,false,true>(temb, TEMPD, m_emb_W + (size_t)i*TEMPD*DMAMB,
                                           m_emb_b + (size_t)i*DMAMB, nullptr,0,
                                           embm, DMAMB, BB, DMAMB, TEMPD, stream);
    k_rmsnorm_add<<<BB*NVV, 256, 0, stream>>>(hm, m_norm_w + (size_t)i*DMAMB,
                                              embm, hn, DMAMB, NVV);
    launch_gemm<EPI_NONE,false,false,true>(hn, DMAMB, m_in_W + (size_t)i*DMAMB*2*DINNER,
                                           nullptr, nullptr,0, xz, 2*DINNER,
                                           BB*NVV, 2*DINNER, DMAMB, stream);
    k_conv_silu<<<(BB*NVV*DINNER+EB-1)/EB, EB, 0, stream>>>(
        xz, m_conv_w + (size_t)i*DINNER*4, m_conv_b + (size_t)i*DINNER, xc);
    launch_gemm<EPI_NONE,false,false,false>(xc, DINNER, m_xproj + (size_t)i*DINNER*112,
                                            nullptr, nullptr,0, dbc, 112,
                                            BB*NVV, 112, DINNER, stream);
    launch_gemm<EPI_SOFTPLUS,false,true,true>(dbc, 112, m_dt_W + (size_t)i*DTRANK*DINNER,
                                              m_dt_b + (size_t)i*DINNER, nullptr,0,
                                              delta, DINNER, BB*NVV, DINNER, DTRANK, stream);
    k_ssm_scan<<<(BB*DINNER+EB-1)/EB, EB, 0, stream>>>(
        delta, dbc, xc, xz, m_A_log + (size_t)i*DINNER*DSTATE, m_D + (size_t)i*DINNER, ymul);
    launch_gemm<EPI_NONE,true,false,true>(ymul, DINNER, m_out_W + (size_t)i*DINNER*DMAMB,
                                          nullptr, hm, DMAMB, hm, DMAMB,
                                          BB*NVV, DMAMB, DINNER, stream);
    k_hm2h<<<(NTOK3+EB-1)/EB, EB, 0, stream>>>(hm, hbuf);
    // ---- attention ----
    k_layernorm<<<BB*TT, 256, 0, stream>>>(hbuf, t_ln1_w + (size_t)i*DTRANS,
                                           t_ln1_b + (size_t)i*DTRANS, xn, DTRANS);
    launch_gemm<EPI_NONE,false,false,true>(xn, DTRANS, t_Wq + (size_t)i*DTRANS*DTRANS,
                                           t_bq + (size_t)i*DTRANS, nullptr,0,
                                           qb, DTRANS, BB*TT, DTRANS, DTRANS, stream);
    launch_gemm<EPI_NONE,false,false,true>(xn, DTRANS, t_Wk + (size_t)i*DTRANS*DTRANS,
                                           t_bk + (size_t)i*DTRANS, nullptr,0,
                                           kb, DTRANS, BB*TT, DTRANS, DTRANS, stream);
    launch_gemm<EPI_NONE,false,false,true>(xn, DTRANS, t_Wv + (size_t)i*DTRANS*DTRANS,
                                           t_bv + (size_t)i*DTRANS, nullptr,0,
                                           vb, DTRANS, BB*TT, DTRANS, DTRANS, stream);
    k_attention<<<BB*NHEADS, 256, 0, stream>>>(qb, kb, vb, yb);
    // ---- stylize 1 (residual into hbuf) ----
    launch_gemm<EPI_NONE,false,false,true>(stemb, TEMPD, t_s1_eW + (size_t)i*TEMPD*2*DTRANS,
                                           t_s1_eb + (size_t)i*2*DTRANS, nullptr,0,
                                           ssb, 2*DTRANS, BB, 2*DTRANS, TEMPD, stream);
    k_ln_style<<<BB*TT, 256, 0, stream>>>(yb, t_s1_lw + (size_t)i*DTRANS,
                                          t_s1_lb + (size_t)i*DTRANS, ssb, sh2, DTRANS);
    launch_gemm<EPI_NONE,true,false,true>(sh2, DTRANS, t_s1_oW + (size_t)i*DTRANS*DTRANS,
                                          t_s1_ob + (size_t)i*DTRANS, hbuf, DTRANS,
                                          hbuf, DTRANS, BB*TT, DTRANS, DTRANS, stream);
    // ---- FFN ----
    k_layernorm<<<BB*TT, 256, 0, stream>>>(hbuf, t_ln2_w + (size_t)i*DTRANS,
                                           t_ln2_b + (size_t)i*DTRANS, xn, DTRANS);
    launch_gemm<EPI_GELU,false,false,true>(xn, DTRANS, t_fW1 + (size_t)i*DTRANS*FFND,
                                           t_fb1 + (size_t)i*FFND, nullptr,0,
                                           f1, FFND, BB*TT, FFND, DTRANS, stream);
    launch_gemm<EPI_NONE,false,false,true>(f1, FFND, t_fW2 + (size_t)i*FFND*DTRANS,
                                           t_fb2 + (size_t)i*DTRANS, nullptr,0,
                                           yb, DTRANS, BB*TT, DTRANS, FFND, stream);
    // ---- stylize 2 (residual into hbuf) ----
    launch_gemm<EPI_NONE,false,false,true>(stemb, TEMPD, t_s2_eW + (size_t)i*TEMPD*2*DTRANS,
                                           t_s2_eb + (size_t)i*2*DTRANS, nullptr,0,
                                           ssb, 2*DTRANS, BB, 2*DTRANS, TEMPD, stream);
    k_ln_style<<<BB*TT, 256, 0, stream>>>(yb, t_s2_lw + (size_t)i*DTRANS,
                                          t_s2_lb + (size_t)i*DTRANS, ssb, sh2, DTRANS);
    launch_gemm<EPI_NONE,true,false,true>(sh2, DTRANS, t_s2_oW + (size_t)i*DTRANS*DTRANS,
                                          t_s2_ob + (size_t)i*DTRANS, hbuf, DTRANS,
                                          hbuf, DTRANS, BB*TT, DTRANS, DTRANS, stream);
    if (i == 2) k_add_ew<<<(NTOK3+EB-1)/EB, EB, 0, stream>>>(hbuf, pre1, NTOK3);
    if (i == 3) k_add_ew<<<(NTOK3+EB-1)/EB, EB, 0, stream>>>(hbuf, pre0, NTOK3);
  }

  // final RMSNorm (over 1920) + REV gather + head projection
  k_rmsnorm_add<<<BB*TT, 256, 0, stream>>>(hbuf, normf_w, nullptr, xn, DTRANS, 1);
  k_head<<<(BB*TT*16*3 + EB-1)/EB, EB, 0, stream>>>(xn, head_W, head_b, out);
}
